// ModifiedGateGRU_13168369729788
// MI455X (gfx1250) — matrix operand order repaired and checked by reading, not compiled, not run on hardware
//
#include <hip/hip_runtime.h>
#include <stdint.h>

#define B_    1024
#define T_    512
#define H_    512
#define DIN_  64
#define DOUT_ 64
#define KC_   576   // H + DIN (combined K)
#define BT_   16    // batch rows per workgroup (one WMMA M tile)

typedef __attribute__((ext_vector_type(16))) __bf16 v16bf;
typedef __attribute__((ext_vector_type(8)))  float  v8f;

static __device__ __forceinline__ uint16_t f2bfu(float f){
  union{float f; unsigned u;} c; c.f = f;
  unsigned r = c.u + 0x7FFFu + ((c.u >> 16) & 1u);   // round-to-nearest-even
  return (uint16_t)(r >> 16);
}
// fast transcendental forms: v_exp_f32 + v_rcp_f32 (no IEEE div sequence)
static __device__ __forceinline__ float sigmf(float x){
  return __builtin_amdgcn_rcpf(1.0f + __expf(-x));
}
static __device__ __forceinline__ float tanhfast(float x){
  return 1.0f - 2.0f*__builtin_amdgcn_rcpf(1.0f + __expf(2.0f*x));
}
static __device__ __forceinline__ v8f zero8(){ v8f z = {0.f,0.f,0.f,0.f,0.f,0.f,0.f,0.f}; return z; }

static __device__ __forceinline__ v16bf ld_frag(const uint16_t* p0, const uint16_t* p1){
  union{ v16bf v; uint4 q[2]; } u;
  u.q[0] = *reinterpret_cast<const uint4*>(p0);
  u.q[1] = *reinterpret_cast<const uint4*>(p1);
  return u.v;
}
static __device__ __forceinline__ v8f wmma_bf16(v16bf a, v16bf b, v8f c){
  return __builtin_amdgcn_wmma_f32_16x16x32_bf16(false, a, false, b, (short)0, c, false, false);
}

// Pack fp32 (K,N) row-major weights into bf16 WMMA B-fragment order:
// tile (kt,nt) of 32x16; lane l holds column n = nt*16+(l&15),
// K = kt*32 + (l>=16 ? 16 : 0) + j for j=0..15, stored contiguously (32B/lane).
__global__ void pack_w_kernel(const float* __restrict__ W, uint16_t* __restrict__ dst,
                              int K, int N){
  int idx = blockIdx.x*blockDim.x + threadIdx.x;
  int total = (K>>5)*(N>>4)*512;
  if (idx >= total) return;
  int j    = idx & 15;
  int l    = (idx >> 4) & 31;
  int tile = idx >> 9;
  int NT   = N >> 4;
  int kt   = tile / NT, nt = tile - kt*NT;
  int n = nt*16 + (l & 15);
  int k = kt*32 + ((l >> 4)*16) + j;
  dst[idx] = f2bfu(W[(size_t)k*N + n]);
}

__global__ __launch_bounds__(256)
void gru_kernel(const float* __restrict__ X0, const float* __restrict__ V,
                const uint16_t* __restrict__ pwr, const uint16_t* __restrict__ pwz,
                const uint16_t* __restrict__ pwh, const uint16_t* __restrict__ pwl,
                const uint16_t* __restrict__ pwo,
                const float* __restrict__ br, const float* __restrict__ bz,
                const float* __restrict__ bh, const float* __restrict__ bl,
                const float* __restrict__ bo,
                float* __restrict__ out)
{
  __shared__ __attribute__((aligned(16))) uint16_t s_comb [BT_*KC_]; // [h_bf16 | v_bf16]
  __shared__ __attribute__((aligned(16))) uint16_t s_combr[BT_*KC_]; // [r*h    | v_bf16]
  __shared__ __attribute__((aligned(16))) float    s_hf   [BT_*H_];  // h fp32
  __shared__ float s_vf[BT_*DIN_];
  __shared__ float s_zo[BT_*DOUT_];
  __shared__ float s_lt[BT_*DOUT_];
  __shared__ float s_xp[BT_*DOUT_];

  const int tid  = threadIdx.x;
  const int lane = tid & 31;
  const int wid  = tid >> 5;
  const int b0   = blockIdx.x * BT_;

  const int cn   = lane & 15;       // column within 16-wide tile (B/C frags)
  const int rsel = lane >> 4;
  const int am   = cn;              // A-fragment row (ISA layout)
  const int akb  = rsel * 16;       // A-fragment K sub-base
  const int cm0  = rsel * 8;        // C-fragment row base

  const int NTg = H_  >> 4;         // 32 n-tiles across H
  const int KTc = KC_ >> 5;         // 18 k-tiles over combined K
  const int KTh = H_  >> 5;         // 16 k-tiles over H
  const int lnt = wid & 3;          // small-matrix n-tile

  // ---- init state: h = 0, x_prev = X0 tile ----
  for (int i = tid; i < BT_*H_; i += 256){
    s_hf[i] = 0.0f;
    s_comb[(i >> 9)*KC_ + (i & 511)] = 0;
  }
  for (int i = tid; i < BT_*DOUT_; i += 256){
    int m = i >> 6, c = i & 63;
    s_xp[i] = X0[(size_t)(b0+m)*DOUT_ + c];
  }
  __syncthreads();

  // per-wave bias values (column depends only on lane&15 and tile index)
  float brv[4], bzv[4], bhv[4];
  #pragma unroll
  for (int j = 0; j < 4; ++j){
    int n = (wid*4+j)*16 + cn;
    brv[j] = br[n]; bzv[j] = bz[n]; bhv[j] = bh[n];
  }
  const float blv = bl[lnt*16 + cn];
  const float bov = bo[lnt*16 + cn];

  for (int t = 0; t < T_; ++t){
    // ---- Phase A: stage v_t (fp32 + bf16 into both combined buffers) ----
    {
      int e0 = tid*4;                       // 16*64 = 1024 = 256 threads * 4
      int m = e0 >> 6, c = e0 & 63;
      const float4 v4 = *reinterpret_cast<const float4*>(
          &V[((size_t)(b0+m)*T_ + t)*DIN_ + c]);
      s_vf[e0+0]=v4.x; s_vf[e0+1]=v4.y; s_vf[e0+2]=v4.z; s_vf[e0+3]=v4.w;
      uint16_t q0=f2bfu(v4.x), q1=f2bfu(v4.y), q2=f2bfu(v4.z), q3=f2bfu(v4.w);
      int base = m*KC_ + H_ + c;
      s_comb [base+0]=q0; s_comb [base+1]=q1; s_comb [base+2]=q2; s_comb [base+3]=q3;
      s_combr[base+0]=q0; s_combr[base+1]=q1; s_combr[base+2]=q2; s_combr[base+3]=q3;
      if (t+1 < T_)
        __builtin_prefetch(&V[((size_t)(b0+m)*T_ + (t+1))*DIN_ + c], 0, 3);
    }
    __syncthreads();

    // ---- Phase B: r, z (this wave's 64 columns) and l, fused K loop ----
    v8f accr[4], accz[4], accl = zero8();
    #pragma unroll
    for (int j = 0; j < 4; ++j){ accr[j] = zero8(); accz[j] = zero8(); }
    for (int kt = 0; kt < KTc; ++kt){
      const uint16_t* ap = s_comb + am*KC_ + kt*32 + akb;
      v16bf af = ld_frag(ap, ap + 8);
      #pragma unroll
      for (int j = 0; j < 4; ++j){
        int ti = (kt*NTg + (wid*4+j))*512 + lane*16;
        v16bf bfr = ld_frag(pwr + ti, pwr + ti + 8);
        v16bf bfz = ld_frag(pwz + ti, pwz + ti + 8);
        accr[j] = wmma_bf16(af, bfr, accr[j]);
        accz[j] = wmma_bf16(af, bfz, accz[j]);
      }
      int li = (kt*4 + lnt)*512 + lane*16;
      v16bf bfl = ld_frag(pwl + li, pwl + li + 8);
      accl = wmma_bf16(af, bfl, accl);
    }
    #pragma unroll
    for (int j = 0; j < 4; ++j){
      int n = (wid*4+j)*16 + cn;
      #pragma unroll
      for (int g = 0; g < 8; ++g){
        int m = cm0 + g;
        float r = sigmf(accr[j][g] + brv[j]);
        float z = sigmf(accz[j][g] + bzv[j]);
        accz[j][g] = z;                                    // keep z in registers
        s_combr[m*KC_ + n] = f2bfu(r * s_hf[m*H_ + n]);    // r*h_prev (bf16)
        if (wid == 0) s_zo[m*DOUT_ + n] = z;               // z[:, :64]
      }
    }
    if (wid < 4){
      #pragma unroll
      for (int g = 0; g < 8; ++g)
        s_lt[(cm0+g)*DOUT_ + lnt*16 + cn] = sigmf(accl[g] + blv);
    }
    __syncthreads();

    // ---- Phase C: h_tilde = tanh([r*h, v] @ W_h + b_h); h update ----
    v8f acch[4];
    #pragma unroll
    for (int j = 0; j < 4; ++j) acch[j] = zero8();
    for (int kt = 0; kt < KTc; ++kt){
      const uint16_t* ap = s_combr + am*KC_ + kt*32 + akb;
      v16bf af = ld_frag(ap, ap + 8);
      #pragma unroll
      for (int j = 0; j < 4; ++j){
        int ti = (kt*NTg + (wid*4+j))*512 + lane*16;
        v16bf bfh = ld_frag(pwh + ti, pwh + ti + 8);
        acch[j] = wmma_bf16(af, bfh, acch[j]);
      }
    }
    #pragma unroll
    for (int j = 0; j < 4; ++j){
      int n = (wid*4+j)*16 + cn;
      #pragma unroll
      for (int g = 0; g < 8; ++g){
        int m = cm0 + g;
        float ht = tanhfast(acch[j][g] + bhv[j]);
        float z  = accz[j][g];
        float hn = (1.0f - z)*s_hf[m*H_ + n] + z*ht;
        s_hf[m*H_ + n]    = hn;
        s_comb[m*KC_ + n] = f2bfu(hn);   // bf16 mirror for next step / W_o GEMM
      }
    }
    __syncthreads();

    // ---- Phase D (waves 0-3): hidden = h @ W_o + b_o; gated output ----
    if (wid < 4){
      v8f acco = zero8();
      for (int kt = 0; kt < KTh; ++kt){
        const uint16_t* ap = s_comb + am*KC_ + kt*32 + akb;
        v16bf af = ld_frag(ap, ap + 8);
        int ti = (kt*4 + wid)*512 + lane*16;
        v16bf bfo = ld_frag(pwo + ti, pwo + ti + 8);
        acco = wmma_bf16(af, bfo, acco);
      }
      int n = wid*16 + cn;
      #pragma unroll
      for (int g = 0; g < 8; ++g){
        int m = cm0 + g;
        float hid = acco[g] + bov;
        float z   = s_zo[m*DOUT_ + n];
        float lin = s_xp[m*DOUT_ + n] + s_lt[m*DOUT_ + n]*s_vf[m*DIN_ + n];
        float x   = (1.0f - z)*lin + z*hid;
        s_xp[m*DOUT_ + n] = x;
        out[((size_t)(b0+m)*T_ + t)*DOUT_ + n] = x;
      }
    }
    __syncthreads();
  }
}

extern "C" void kernel_launch(void* const* d_in, const int* in_sizes, int n_in,
                              void* d_out, int out_size, void* d_ws, size_t ws_size,
                              hipStream_t stream){
  const float* X0 = (const float*)d_in[0];
  const float* V  = (const float*)d_in[1];
  const float* Wr = (const float*)d_in[2];
  const float* br = (const float*)d_in[3];
  const float* Wz = (const float*)d_in[4];
  const float* bz = (const float*)d_in[5];
  const float* Wh = (const float*)d_in[6];
  const float* bh = (const float*)d_in[7];
  const float* Wl = (const float*)d_in[8];
  const float* bl = (const float*)d_in[9];
  const float* Wo = (const float*)d_in[10];
  const float* bo = (const float*)d_in[11];
  float* out = (float*)d_out;

  uint16_t* ws = (uint16_t*)d_ws;
  const size_t szg = (size_t)(KC_/32)*(H_/16)*512;    // 294912 elems per gate weight
  const size_t szl = (size_t)(KC_/32)*(DOUT_/16)*512; // 36864
  const size_t szo = (size_t)(H_/32) *(DOUT_/16)*512; // 32768
  uint16_t* pr = ws;
  uint16_t* pz = pr + szg;
  uint16_t* ph = pz + szg;
  uint16_t* pl = ph + szg;
  uint16_t* po = pl + szl;

  pack_w_kernel<<<(int)((szg+255)/256), 256, 0, stream>>>(Wr, pr, KC_, H_);
  pack_w_kernel<<<(int)((szg+255)/256), 256, 0, stream>>>(Wz, pz, KC_, H_);
  pack_w_kernel<<<(int)((szg+255)/256), 256, 0, stream>>>(Wh, ph, KC_, H_);
  pack_w_kernel<<<(int)((szl+255)/256), 256, 0, stream>>>(Wl, pl, KC_, DOUT_);
  pack_w_kernel<<<(int)((szo+255)/256), 256, 0, stream>>>(Wo, po, H_,  DOUT_);

  gru_kernel<<<B_/BT_, 256, 0, stream>>>(X0, V, pr, pz, ph, pl, po,
                                         br, bz, bh, bl, bo, out);
}
